// GraphNet_75479755260068
// MI455X (gfx1250) — compile-verified, matrix operand order checked by reading, not run on hardware
//
#include <hip/hip_runtime.h>
#include <stdint.h>

#define NN 50000
#define NE 800000
#define NG 256

typedef __attribute__((ext_vector_type(16))) __bf16 v16bf;
typedef __attribute__((ext_vector_type(16))) unsigned short v16u;
typedef __attribute__((ext_vector_type(8)))  float v8f;

union BF16x16 { v16u u; v16bf bf; };

__device__ __forceinline__ unsigned short f2bf(float f) {
  unsigned u = __float_as_uint(f);
  u += 0x7FFFu + ((u >> 16) & 1u);          // round-to-nearest-even
  return (unsigned short)(u >> 16);
}

// ---- pack W[K,F] (fp32, row-major) into WMMA-B layout (bf16) ----
// Wb[((kt*F + col) * 32) + kk] = bf16(W[kt*32+kk][col]), zero-padded past KACT.
template<int KACT, int F>
__global__ void pack_w(const float* __restrict__ W, unsigned short* __restrict__ Wb) {
  constexpr int KT = (KACT + 31) / 32;
  int idx = blockIdx.x * 256 + threadIdx.x;
  if (idx >= KT * 32 * F) return;
  int kk   = idx & 31;
  int rest = idx >> 5;          // kt*F + col
  int col  = rest % F;
  int kt   = rest / F;
  int k    = kt * 32 + kk;
  float v  = (k < KACT) ? W[(size_t)k * F + col] : 0.0f;
  Wb[idx] = f2bf(v);
}

// ---- pack activations [N, KACT] fp32 -> [N, KPAD] bf16 (relu fused, zero pad) ----
// Each thread emits 4 consecutive bf16 (one 8-byte store).
template<int KACT, int KPAD, bool RELU>
__global__ void pack_a(const float* __restrict__ A, unsigned short* __restrict__ Abf) {
  constexpr int C = KPAD / 4;
  int t = blockIdx.x * 256 + threadIdx.x;
  if (t >= NN * C) return;
  int i  = t / C;
  int k0 = (t - i * C) * 4;
  const float* row = A + (size_t)i * KACT;
  union { unsigned short h[4]; uint2 u2; } o;
#pragma unroll
  for (int j = 0; j < 4; ++j) {
    float v = 0.0f;
    if ((KACT % 4 == 0 && KPAD == KACT) || (k0 + j < KACT)) v = row[k0 + j];
    if (RELU) v = fmaxf(v, 0.0f);
    o.h[j] = f2bf(v);
  }
  *(uint2*)(Abf + (size_t)i * KPAD + k0) = o.u2;
}

// ---- WMMA GEMM: out[N,F] = Abf[N, KT*32] @ Wb ----
// block = 128 threads (4 waves); wave w -> 16x16 tile (blockIdx.x rows, col tile blockIdx.y*4+w)
template<int KT, int F>
__global__ __launch_bounds__(128) void gemm_wmma(const unsigned short* __restrict__ Abf,
                                                 const unsigned short* __restrict__ Wb,
                                                 float* __restrict__ out) {
  constexpr int KPAD = KT * 32;
  const int lane  = threadIdx.x & 31;
  const int wave  = threadIdx.x >> 5;
  const int rowBase = blockIdx.x << 4;
  const int col0    = ((blockIdx.y << 2) + wave) << 4;
  const int row   = rowBase + (lane & 15);
  const int colB  = col0 + (lane & 15);
  const int khalf = (lane >> 4) << 4;          // 0 or 16

  const unsigned short* ap = Abf + (size_t)row * KPAD + khalf;
  const unsigned short* bp = Wb + (((size_t)colB) << 5) + khalf;

  v8f acc = {};
#pragma unroll
  for (int kt = 0; kt < KT; ++kt) {
    BF16x16 a, b;
    a.u = *(const v16u*)(ap + kt * 32);
    b.u = *(const v16u*)(bp + (size_t)kt * F * 32);
    acc = __builtin_amdgcn_wmma_f32_16x16x32_bf16(false, a.bf, false, b.bf,
                                                  (short)0, acc, false, false);
  }
  const int orow = rowBase + ((lane >> 4) << 3);
  float* o = out + (size_t)orow * F + colB;
#pragma unroll
  for (int r = 0; r < 8; ++r) o[(size_t)r * F] = acc[r];
}

// ---- degree / norm ----
__global__ void deg_init(int* deg) {
  int i = blockIdx.x * 256 + threadIdx.x;
  if (i < NN) deg[i] = 1;                       // self loop
}
__global__ void deg_count(const int* __restrict__ dst, int* deg) {
  int e = blockIdx.x * 256 + threadIdx.x;
  if (e < NE) atomicAdd(&deg[dst[e]], 1);
}
__global__ void dinv_k(const int* __restrict__ deg, float* __restrict__ dinv) {
  int i = blockIdx.x * 256 + threadIdx.x;
  if (i < NN) dinv[i] = rsqrtf((float)deg[i]);
}
__global__ void enorm_k(const int* __restrict__ src, const int* __restrict__ dst,
                        const float* __restrict__ dinv, float* __restrict__ enorm) {
  int e = blockIdx.x * 256 + threadIdx.x;
  if (e < NE) enorm[e] = dinv[src[e]] * dinv[dst[e]];
}

// ---- out[i,f] = h[i,f]*dinv[i]^2 + b[f]  (self-loop term + bias init) ----
template<int F>
__global__ void selfloop_bias(const float* __restrict__ h, const float* __restrict__ dinv,
                              const float* __restrict__ b, float* __restrict__ out) {
  int idx = blockIdx.x * 256 + threadIdx.x;
  if (idx >= NN * F) return;
  int i = idx / F;
  int f = idx & (F - 1);
  float di = dinv[i];
  out[idx] = h[idx] * di * di + b[f];
}

// ---- scatter-add:  out[dst] += h[src] * enorm[e]  (float4 chunks) ----
template<int F>
__global__ void edge_agg(const int* __restrict__ src, const int* __restrict__ dst,
                         const float* __restrict__ enorm, const float* __restrict__ h,
                         float* __restrict__ out) {
  constexpr int F4 = F / 4;
  int t = blockIdx.x * 256 + threadIdx.x;
  if (t >= NE * F4) return;
  int e = t / F4;
  int c = t & (F4 - 1);
  int s = src[e], d = dst[e];
  float nrm = enorm[e];
  float4 hv = ((const float4*)(h + (size_t)s * F))[c];
  float* o = out + (size_t)d * F + (c << 2);
  atomicAdd(o + 0, hv.x * nrm);
  atomicAdd(o + 1, hv.y * nrm);
  atomicAdd(o + 2, hv.z * nrm);
  atomicAdd(o + 3, hv.w * nrm);
}

// ---- global max pool (fused relu; non-neg floats compare as ints) ----
__global__ void pool_init(int* pooled) {
  int i = blockIdx.x * 256 + threadIdx.x;
  if (i < NG * 256) pooled[i] = 0;              // 0.0f bits
}
__global__ void pool_max(const float* __restrict__ h3, const int* __restrict__ batch,
                         int* __restrict__ pooled) {
  int t = blockIdx.x * 256 + threadIdx.x;
  if (t >= NN * 64) return;
  int i = t >> 6;
  int c = t & 63;
  int g = batch[i];
  float4 v = ((const float4*)(h3 + (size_t)i * 256))[c];
  int* p = pooled + g * 256 + (c << 2);
  atomicMax(p + 0, __float_as_int(fmaxf(v.x, 0.0f)));
  atomicMax(p + 1, __float_as_int(fmaxf(v.y, 0.0f)));
  atomicMax(p + 2, __float_as_int(fmaxf(v.z, 0.0f)));
  atomicMax(p + 3, __float_as_int(fmaxf(v.w, 0.0f)));
}

// ---- head: z = BN(relu(pooled@Wl1+bl1)); out = z@Wl2+bl2 ----
__global__ __launch_bounds__(128) void head_k(const int* __restrict__ pooled_bits,
                                              const float* __restrict__ Wl1, const float* __restrict__ bl1,
                                              const float* __restrict__ gamma, const float* __restrict__ beta,
                                              const float* __restrict__ rmean, const float* __restrict__ rvar,
                                              const float* __restrict__ Wl2, const float* __restrict__ bl2,
                                              float* __restrict__ out) {
  __shared__ float red[128];
  int g = blockIdx.x, j = threadIdx.x;
  const float* p = (const float*)pooled_bits + (size_t)g * 256;
  float acc = bl1[j];
#pragma unroll 4
  for (int k = 0; k < 256; ++k) acc += p[k] * Wl1[(size_t)k * 128 + j];
  float z = fmaxf(acc, 0.0f);
  z = (z - rmean[j]) * rsqrtf(rvar[j] + 1e-5f) * gamma[j] + beta[j];
  red[j] = z * Wl2[j];
  __syncthreads();
  for (int s = 64; s > 0; s >>= 1) {
    if (j < s) red[j] += red[j + s];
    __syncthreads();
  }
  if (j == 0) out[g] = red[0] + bl2[0];
}

static inline int cdiv(long long a, int b) { return (int)((a + b - 1) / b); }

extern "C" void kernel_launch(void* const* d_in, const int* in_sizes, int n_in,
                              void* d_out, int out_size, void* d_ws, size_t ws_size,
                              hipStream_t stream) {
  const float* x    = (const float*)d_in[0];
  const int*   ei   = (const int*)d_in[1];
  const int*   src  = ei;
  const int*   dst  = ei + NE;
  const int*   batch= (const int*)d_in[2];
  const float* W1 = (const float*)d_in[3];  const float* b1 = (const float*)d_in[4];
  const float* W2 = (const float*)d_in[5];  const float* b2 = (const float*)d_in[6];
  const float* W3 = (const float*)d_in[7];  const float* b3 = (const float*)d_in[8];
  const float* Wl1= (const float*)d_in[9];  const float* bl1= (const float*)d_in[10];
  const float* gam= (const float*)d_in[11]; const float* bet= (const float*)d_in[12];
  const float* rm = (const float*)d_in[13]; const float* rv = (const float*)d_in[14];
  const float* Wl2= (const float*)d_in[15]; const float* bl2= (const float*)d_in[16];
  float* out = (float*)d_out;

  // workspace carve (256B aligned)
  char* p = (char*)d_ws;
  auto carve = [&](size_t bytes) -> char* {
    char* r = (char*)(((uintptr_t)p + 255) & ~(uintptr_t)255);
    p = r + bytes;
    return r;
  };
  float* bufA = (float*)carve((size_t)NN * 256 * 4);            // transformed h (fp32)
  float* bufB = (float*)carve((size_t)NN * 256 * 4);            // aggregated (fp32)
  unsigned short* bufC = (unsigned short*)carve((size_t)NN * 128 * 2); // packed bf16 A
  int*   deg  = (int*)  carve((size_t)NN * 4);
  float* dinv = (float*)carve((size_t)NN * 4);
  float* enrm = (float*)carve((size_t)NE * 4);
  int*   pool = (int*)  carve((size_t)NG * 256 * 4);
  unsigned short* Wb1 = (unsigned short*)carve((size_t)64  * 64  * 2);  // K padded 64
  unsigned short* Wb2 = (unsigned short*)carve((size_t)64  * 128 * 2);
  unsigned short* Wb3 = (unsigned short*)carve((size_t)128 * 256 * 2);

  // weight packing (tiny)
  pack_w<37, 64 ><<<cdiv(2 * 32 * 64, 256),  256, 0, stream>>>(W1, Wb1);
  pack_w<64, 128><<<cdiv(2 * 32 * 128, 256), 256, 0, stream>>>(W2, Wb2);
  pack_w<128,256><<<cdiv(4 * 32 * 256, 256), 256, 0, stream>>>(W3, Wb3);

  // degrees / norms
  deg_init <<<cdiv(NN, 256), 256, 0, stream>>>(deg);
  deg_count<<<cdiv(NE, 256), 256, 0, stream>>>(dst, deg);
  dinv_k   <<<cdiv(NN, 256), 256, 0, stream>>>(deg, dinv);
  enorm_k  <<<cdiv(NE, 256), 256, 0, stream>>>(src, dst, dinv, enrm);
  pool_init<<<cdiv(NG * 256, 256), 256, 0, stream>>>(pool);

  const int MT = NN / 16;   // 3125 exact

  // layer 1: pack x (pad 37->64), h = x@W1, o = self+bias then scatter
  pack_a<37, 64, false><<<cdiv((long long)NN * 16, 256), 256, 0, stream>>>(x, bufC);
  gemm_wmma<2, 64><<<dim3(MT, 1), 128, 0, stream>>>(bufC, Wb1, bufA);
  selfloop_bias<64><<<cdiv((long long)NN * 64, 256), 256, 0, stream>>>(bufA, dinv, b1, bufB);
  edge_agg<64><<<cdiv((long long)NE * 16, 256), 256, 0, stream>>>(src, dst, enrm, bufA, bufB);

  // layer 2 (relu fused into pack)
  pack_a<64, 64, true><<<cdiv((long long)NN * 16, 256), 256, 0, stream>>>(bufB, bufC);
  gemm_wmma<2, 128><<<dim3(MT, 2), 128, 0, stream>>>(bufC, Wb2, bufA);
  selfloop_bias<128><<<cdiv((long long)NN * 128, 256), 256, 0, stream>>>(bufA, dinv, b2, bufB);
  edge_agg<128><<<cdiv((long long)NE * 32, 256), 256, 0, stream>>>(src, dst, enrm, bufA, bufB);

  // layer 3
  pack_a<128, 128, true><<<cdiv((long long)NN * 32, 256), 256, 0, stream>>>(bufB, bufC);
  gemm_wmma<4, 256><<<dim3(MT, 4), 128, 0, stream>>>(bufC, Wb3, bufA);
  selfloop_bias<256><<<cdiv((long long)NN * 256, 256), 256, 0, stream>>>(bufA, dinv, b3, bufB);
  edge_agg<256><<<cdiv((long long)NE * 64, 256), 256, 0, stream>>>(src, dst, enrm, bufA, bufB);

  // pool (fused relu) + head
  pool_max<<<cdiv((long long)NN * 64, 256), 256, 0, stream>>>(bufB, batch, pool);
  head_k<<<NG, 128, 0, stream>>>(pool, Wl1, bl1, gam, bet, rm, rv, Wl2, bl2, out);
}
